// VectorQuantizer_10969346474532
// MI455X (gfx1250) — matrix operand order repaired and checked by reading, not compiled, not run on hardware
//
#include <hip/hip_runtime.h>

typedef __attribute__((ext_vector_type(16))) __bf16 v16bf;
typedef __attribute__((ext_vector_type(8)))  __bf16 v8bf;
typedef __attribute__((ext_vector_type(8)))  float  v8f;

constexpr int C_DIM   = 64;     // HIDDEN_DIM
constexpr int K_CODES = 512;    // N_EMBEDS
constexpr int HW      = 1024;   // 32*32 spatial, channel stride in x / out
constexpr int PTS_PER_BLOCK = 128;
constexpr int THREADS = 256;    // 8 wave32

// Dynamic-LDS layout (bytes)
constexpr int OFF_EHI = 0;                              // K*C bf16 = 65536
constexpr int OFF_ELO = OFF_EHI + K_CODES * C_DIM * 2;  // 65536
constexpr int OFF_ESQ = OFF_ELO + K_CODES * C_DIM * 2;  // K*4 = 2048
constexpr int OFF_IDX = OFF_ESQ + K_CODES * 4;          // 128*4
constexpr int SMEM_BYTES = OFF_IDX + PTS_PER_BLOCK * 4; // 133632 B < 320 KB

static __device__ inline v8f wmma_bf16(v16bf a, v16bf b, v8f c) {
  // v_wmma_f32_16x16x32_bf16: D = A(16x32) * B(32x16) + C(16x16 f32)
  return __builtin_amdgcn_wmma_f32_16x16x32_bf16(
      /*neg_a=*/false, a, /*neg_b=*/false, b,
      /*c_mod=*/(short)0, c, /*reuse_a=*/false, /*reuse_b=*/false);
}

static __device__ inline v16bf cat8(v8bf a, v8bf b) {
  return __builtin_shufflevector(a, b, 0, 1, 2, 3, 4, 5, 6, 7,
                                       8, 9, 10, 11, 12, 13, 14, 15);
}

// One code-tile's A operands: hi/lo split, 2 k-steps of 32 channels.
struct ATile {
  v16bf h[2];
  v16bf l[2];
};

static __device__ inline ATile load_atile(const __bf16* __restrict__ Ehi,
                                          const __bf16* __restrict__ Elo,
                                          int ct, int lp, int half) {
  ATile t;
  const int codeRow = ct * 16 + lp;   // A: M = lane&15
#pragma unroll
  for (int s = 0; s < 2; ++s) {
    // A 16-bit layout: lane half selects Kb=0/8; elems 0-7 = K=Kb..Kb+7,
    // elems 8-15 = K=16+Kb..16+Kb+7 (within this 32-wide k-step).
    const int eidx = codeRow * C_DIM + s * 32 + half * 8;
    const v8bf* ph = (const v8bf*)(Ehi + eidx);
    const v8bf* pl = (const v8bf*)(Elo + eidx);
    t.h[s] = cat8(ph[0], ph[2]);  // +16 bf16 = +2 v8bf
    t.l[s] = cat8(pl[0], pl[2]);
  }
  return t;
}

__global__ __launch_bounds__(THREADS)
void vq_argmin_gather_kernel(const float* __restrict__ x,
                             const float* __restrict__ ew,
                             float* __restrict__ out) {
  extern __shared__ char smem[];
  __bf16* Ehi  = (__bf16*)(smem + OFF_EHI);
  __bf16* Elo  = (__bf16*)(smem + OFF_ELO);
  float*  esq  = (float*)(smem + OFF_ESQ);
  int*    sidx = (int*)(smem + OFF_IDX);

  const int tid = threadIdx.x;

  // ---- Stage codebook into LDS as bf16 hi/lo split ----
  for (int t = tid; t < K_CODES * C_DIM; t += THREADS) {
    float f = ew[t];
    __bf16 h = (__bf16)f;
    Ehi[t] = h;
    Elo[t] = (__bf16)(f - (float)h);
  }
  // ---- e_sq[k] = sum_c e[k][c]^2 (fp32) ----
  for (int k = tid; k < K_CODES; k += THREADS) {
    const float* r = ew + k * C_DIM;
    float s = 0.f;
#pragma unroll 8
    for (int c = 0; c < C_DIM; ++c) s += r[c] * r[c];
    esq[k] = s;
  }
  __syncthreads();

  const int wave = tid >> 5;
  const int lane = tid & 31;
  const int half = lane >> 4;   // 0: lanes 0-15, 1: lanes 16-31
  const int lp   = lane & 15;

  // This lane's point (N dim of WMMA), 16 points per wave.
  const int  n_pt = blockIdx.x * PTS_PER_BLOCK + wave * 16 + lp;
  const long xrow = (long)(n_pt >> 10) * (C_DIM * HW) + (n_pt & (HW - 1));

  // ---- B operand: x^T tile, bf16 hi/lo, 2 k-steps of 32 channels.
  // B layout: lane half selects Kb=0/8; elems 0-7 = K=Kb..Kb+7, elems 8-15 = K=16+Kb..16+Kb+7.
  v16bf Xh[2], Xl[2];
#pragma unroll
  for (int s = 0; s < 2; ++s) {
#pragma unroll
    for (int j = 0; j < 16; ++j) {
      int c = s * 32 + (j & 8) * 2 + half * 8 + (j & 7);
      float f = x[xrow + (long)c * HW];
      __bf16 h = (__bf16)f;
      Xh[s][j] = h;
      Xl[s][j] = (__bf16)(f - (float)h);
    }
  }

  float minv = 3.4e38f;
  int   mini = 0;

  // ---- Sweep 32 code tiles of 16 codes (M dim of WMMA), software-pipelined:
  // A tile for ct+1 is loaded from LDS while tile ct's WMMAs + argmin run.
  ATile cur = load_atile(Ehi, Elo, 0, lp, half);
#pragma unroll 1
  for (int ct = 0; ct < K_CODES / 16; ++ct) {
    ATile nxt = load_atile(Ehi, Elo, (ct + 1) & (K_CODES / 16 - 1), lp, half);

    v8f acc = {};
#pragma unroll
    for (int s = 0; s < 2; ++s) {
      // cross ~= Eh*Xh + Eh*Xl + El*Xh  (fp32-class accuracy)
      acc = wmma_bf16(cur.h[s], Xh[s], acc);
      acc = wmma_bf16(cur.h[s], Xl[s], acc);
      acc = wmma_bf16(cur.l[s], Xh[s], acc);
    }

    // D layout: VGPR r -> code = ct*16 + half*8 + r for this lane's point.
    const float* eq = esq + ct * 16 + half * 8;
#pragma unroll
    for (int r = 0; r < 8; ++r) {
      float d = eq[r] - 2.0f * acc[r];
      int code = ct * 16 + half * 8 + r;
      if (d < minv) { minv = d; mini = code; }
    }
    cur = nxt;
  }

  // Combine the two half-wave partials for each point (lanes l and l^16).
  float ov = __shfl_xor(minv, 16, 32);
  int   oi = __shfl_xor(mini, 16, 32);
  if (ov < minv || (ov == minv && oi < mini)) { minv = ov; mini = oi; }
  if (half == 0) sidx[wave * 16 + lp] = mini;
  __syncthreads();

  // ---- Gather + transposed write-back: out[b][c][hw] = E[idx[n]][c] ----
  const int nb  = blockIdx.x * PTS_PER_BLOCK;   // 128 | 1024 -> single b
  const int b   = nb >> 10;
  const int hw0 = nb & (HW - 1);
  const int i     = tid & 127;          // point within block
  const int cbase = (tid >> 7) * 32;    // two channel halves across the block
  const int code  = sidx[i];
  float* obase = out + (long)b * (C_DIM * HW) + hw0 + i;
  const float* erow = ew + code * C_DIM;
#pragma unroll 4
  for (int c = 0; c < 32; ++c) {
    int cc = cbase + c;
    obase[(long)cc * HW] = erow[cc];    // exact fp32 codebook values
  }
}

extern "C" void kernel_launch(void* const* d_in, const int* in_sizes, int n_in,
                              void* d_out, int out_size, void* d_ws, size_t ws_size,
                              hipStream_t stream) {
  (void)n_in; (void)d_ws; (void)ws_size; (void)out_size;
  const float* x  = (const float*)d_in[0];   // (64, 64, 32, 32) fp32
  const float* ew = (const float*)d_in[1];   // (512, 64) fp32
  float* out = (float*)d_out;                // (64, 64, 32, 32) fp32

  const int n_points = in_sizes[0] / C_DIM;          // 65536
  const int grid     = n_points / PTS_PER_BLOCK;     // 512 blocks
  vq_argmin_gather_kernel<<<grid, THREADS, SMEM_BYTES, stream>>>(x, ew, out);
}
